// ECGGNNV2_6330781794759
// MI455X (gfx1250) — matrix operand order repaired and checked by reading, not compiled, not run on hardware
//
#include <hip/hip_runtime.h>
#include <hip/hip_bf16.h>
#include <stdint.h>

// ---------------- problem constants (from reference) ----------------
#define NN   50000      // nodes
#define NE   400000     // edges
#define ND   128        // node dim
#define ED   192        // edge dim
#define HID  256        // hidden

// ---------------- WMMA types ----------------
typedef __bf16  v16bf __attribute__((ext_vector_type(16)));
typedef float   v8f   __attribute__((ext_vector_type(8)));

#define WMMA_BF16(a, b, c) \
  __builtin_amdgcn_wmma_f32_16x16x32_bf16(false, (a), false, (b), (short)0, (c), false, false)

union Frag {
  uint32_t u[8];
  uint4    q[2];
  v16bf    v;
};

// f32 -> bf16 (RNE) pair pack, used only in the weight-prep kernel
__device__ __forceinline__ uint32_t pk2bf(float a, float b) {
  uint32_t ua = __float_as_uint(a);
  uint32_t ub = __float_as_uint(b);
  ua += 0x7FFFu + ((ua >> 16) & 1u);
  ub += 0x7FFFu + ((ub >> 16) & 1u);
  return (ua >> 16) | (ub & 0xFFFF0000u);
}

__device__ __forceinline__ float mish_f(float x) {
  float sp = (x > 20.f) ? x : __logf(1.f + __expf(x));
  return x * tanhf(sp);
}

// A fragment (16x32 bf16) straight from a bf16 row in LDS (row = lane&15).
// ISA layout: lanes 0-15 hold K {0..7, 16..23}; lanes 16-31 hold K {8..15, 24..31}.
__device__ __forceinline__ v16bf make_a_frag(const __bf16* rowp, int k0, int lane) {
  const __bf16* p = rowp + k0 + ((lane >> 4) << 3);
  Frag fr;
  fr.q[0] = *(const uint4*)(p);
  fr.q[1] = *(const uint4*)(p + 16);
  return fr.v;
}

// B fragment (32x16 bf16) from packed-transposed weights Wpk[n][kk] (kk = K/2 pairs).
// ISA layout: lanes 0-15 hold K {0..7, 16..23} of the chunk (col N = lane), lanes 16-31 hold K {8..15, 24..31}.
__device__ __forceinline__ v16bf make_b_frag(const uint32_t* __restrict__ wpk,
                                             int kkStride, int ntile, int kchunk, int lane) {
  const uint32_t* p = wpk + (size_t)((ntile << 4) + (lane & 15)) * kkStride
                          + (kchunk << 4) + ((lane >> 4) << 2);
  Frag fr;
  fr.q[0] = *(const uint4*)(p);
  fr.q[1] = *(const uint4*)(p + 8);
  return fr.v;
}

__device__ __forceinline__ v8f bias_acc(float bv) {
  v8f acc;
#pragma unroll
  for (int i = 0; i < 8; ++i) acc[i] = bv;
  return acc;
}

// ---------------- prep: pack weights W[K][N] fp32 -> Wpk[n][kk] bf16-pair dwords --------
__global__ __launch_bounds__(256) void pack_w_kernel(const float* __restrict__ W,
                                                     uint32_t* __restrict__ out,
                                                     int K2, int N) {
  int i = blockIdx.x * 256 + threadIdx.x;
  if (i >= N * K2) return;
  int n  = i / K2;
  int kk = i - n * K2;
  float a = W[(size_t)(2 * kk)     * N + n];
  float b = W[(size_t)(2 * kk + 1) * N + n];
  out[(size_t)n * K2 + kk] = pk2bf(a, b);
}

__global__ __launch_bounds__(256) void zero_f32_kernel(float4* __restrict__ p, int n4) {
  int i = blockIdx.x * 256 + threadIdx.x;
  if (i < n4) p[i] = make_float4(0.f, 0.f, 0.f, 0.f);
}

// ---------------- edge message + scatter-add ----------------
// block = 256 threads (8 waves), 64 edges (4 M-tiles), 2 col-tiles per wave.
// K-chunk-outer loop: 4 A-fragments loaded once per kc, applied to both B tiles
// (halves A-LDS traffic per WMMA). Static LDS ~90.6 KB.
__global__ __launch_bounds__(256) void edge_msg_kernel(
    const float* __restrict__ x, const long long* __restrict__ eidx,
    const float* __restrict__ eattr,
    const float* __restrict__ b1, const float* __restrict__ b2,
    const uint32_t* __restrict__ w1pk, const uint32_t* __restrict__ w2pk,
    float* __restrict__ agg) {
  __shared__ __bf16 sIn[64 * 448];   // concat input, bf16 (56 KB)
  __shared__ __bf16 sH[64 * 256];    // mish(layer1), bf16 (32 KB)
  __shared__ int    sDst[64];
  __shared__ int    sSrc[64];

  const int tid = threadIdx.x;
  const long long eBase = (long long)blockIdx.x * 64;

  if (tid < 64) {
    sSrc[tid] = (int)eidx[eBase + tid];
    sDst[tid] = (int)eidx[(long long)NE + eBase + tid];
  }
  __syncthreads();

  // gather stage: [x_dst(128) | x_src(128) | edge_attr(192)] per edge row, cvt to bf16 once
  for (int idx = tid; idx < 64 * 448; idx += 256) {
    int r = idx / 448;
    int c = idx - r * 448;
    float v;
    if (c < 128)       v = x[(size_t)sDst[r] * ND + c];
    else if (c < 256)  v = x[(size_t)sSrc[r] * ND + (c - 128)];
    else               v = eattr[(eBase + r) * ED + (c - 256)];
    sIn[idx] = (__bf16)v;
  }
  __syncthreads();

  const int wv = tid >> 5;
  const int lane = tid & 31;
  const int col = lane & 15;
  const int rb = (lane >> 4) << 3;
  const int arow = lane & 15;
  const int t0 = wv, t1 = wv + 8;

  // layer 1: 64x448 @ 448x256
  {
    const float bv0 = b1[(t0 << 4) + col];
    const float bv1 = b1[(t1 << 4) + col];
    v8f acc[2][4] = {{bias_acc(bv0), bias_acc(bv0), bias_acc(bv0), bias_acc(bv0)},
                     {bias_acc(bv1), bias_acc(bv1), bias_acc(bv1), bias_acc(bv1)}};
#pragma unroll
    for (int kc = 0; kc < 14; ++kc) {
      v16bf a0 = make_a_frag(sIn + (size_t)(arow)      * 448, kc << 5, lane);
      v16bf a1 = make_a_frag(sIn + (size_t)(16 + arow) * 448, kc << 5, lane);
      v16bf a2 = make_a_frag(sIn + (size_t)(32 + arow) * 448, kc << 5, lane);
      v16bf a3 = make_a_frag(sIn + (size_t)(48 + arow) * 448, kc << 5, lane);
      v16bf bf0 = make_b_frag(w1pk, 224, t0, kc, lane);
      acc[0][0] = WMMA_BF16(a0, bf0, acc[0][0]);
      acc[0][1] = WMMA_BF16(a1, bf0, acc[0][1]);
      acc[0][2] = WMMA_BF16(a2, bf0, acc[0][2]);
      acc[0][3] = WMMA_BF16(a3, bf0, acc[0][3]);
      v16bf bf1 = make_b_frag(w1pk, 224, t1, kc, lane);
      acc[1][0] = WMMA_BF16(a0, bf1, acc[1][0]);
      acc[1][1] = WMMA_BF16(a1, bf1, acc[1][1]);
      acc[1][2] = WMMA_BF16(a2, bf1, acc[1][2]);
      acc[1][3] = WMMA_BF16(a3, bf1, acc[1][3]);
    }
#pragma unroll
    for (int tt = 0; tt < 2; ++tt) {
      const int t = wv + (tt << 3);
#pragma unroll
      for (int m = 0; m < 4; ++m)
#pragma unroll
        for (int i = 0; i < 8; ++i)
          sH[((m << 4) + rb + i) * 256 + (t << 4) + col] = (__bf16)mish_f(acc[tt][m][i]);
    }
  }
  __syncthreads();

  // layer 2: 64x256 @ 256x256, then atomic scatter-add to agg[dst]
  {
    const float bv0 = b2[(t0 << 4) + col];
    const float bv1 = b2[(t1 << 4) + col];
    v8f acc[2][4] = {{bias_acc(bv0), bias_acc(bv0), bias_acc(bv0), bias_acc(bv0)},
                     {bias_acc(bv1), bias_acc(bv1), bias_acc(bv1), bias_acc(bv1)}};
#pragma unroll
    for (int kc = 0; kc < 8; ++kc) {
      v16bf a0 = make_a_frag(sH + (size_t)(arow)      * 256, kc << 5, lane);
      v16bf a1 = make_a_frag(sH + (size_t)(16 + arow) * 256, kc << 5, lane);
      v16bf a2 = make_a_frag(sH + (size_t)(32 + arow) * 256, kc << 5, lane);
      v16bf a3 = make_a_frag(sH + (size_t)(48 + arow) * 256, kc << 5, lane);
      v16bf bf0 = make_b_frag(w2pk, 128, t0, kc, lane);
      acc[0][0] = WMMA_BF16(a0, bf0, acc[0][0]);
      acc[0][1] = WMMA_BF16(a1, bf0, acc[0][1]);
      acc[0][2] = WMMA_BF16(a2, bf0, acc[0][2]);
      acc[0][3] = WMMA_BF16(a3, bf0, acc[0][3]);
      v16bf bf1 = make_b_frag(w2pk, 128, t1, kc, lane);
      acc[1][0] = WMMA_BF16(a0, bf1, acc[1][0]);
      acc[1][1] = WMMA_BF16(a1, bf1, acc[1][1]);
      acc[1][2] = WMMA_BF16(a2, bf1, acc[1][2]);
      acc[1][3] = WMMA_BF16(a3, bf1, acc[1][3]);
    }
#pragma unroll
    for (int tt = 0; tt < 2; ++tt) {
      const int t = wv + (tt << 3);
#pragma unroll
      for (int m = 0; m < 4; ++m)
#pragma unroll
        for (int i = 0; i < 8; ++i) {
          const int d = sDst[(m << 4) + rb + i];
          unsafeAtomicAdd(&agg[(size_t)d * HID + (t << 4) + col], acc[tt][m][i]);
        }
    }
  }
}

// ---------------- node update + residual + layernorm ----------------
// block = 256 threads (8 waves), 64 nodes (4 M-tiles). Tail block handled by
// index clamping (staging) + store guards; WMMA path stays divergence-free.
__global__ __launch_bounds__(256) void node_upd_kernel(
    const float* __restrict__ x, const float* __restrict__ agg,
    const float* __restrict__ c1, const float* __restrict__ c2,
    const uint32_t* __restrict__ u1pk, const uint32_t* __restrict__ u2pk,
    const float* __restrict__ gamma, const float* __restrict__ beta,
    float* __restrict__ out) {
  __shared__ __bf16 sIn[64 * 384];   // 48 KB
  __shared__ __bf16 sH[64 * 256];    // 32 KB
  __shared__ float  sU[64 * 128];    // 32 KB

  const int tid = threadIdx.x;
  const long long nBase = (long long)blockIdx.x * 64;

  for (int idx = tid; idx < 64 * 384; idx += 256) {
    int r = idx / 384;
    int c = idx - r * 384;
    long long row = nBase + r;
    if (row > NN - 1) row = NN - 1;   // clamp tail (results discarded at store)
    float v = (c < 128) ? x[row * ND + c] : agg[row * HID + (c - 128)];
    sIn[idx] = (__bf16)v;
  }
  __syncthreads();

  const int wv = tid >> 5;
  const int lane = tid & 31;
  const int col = lane & 15;
  const int rb = (lane >> 4) << 3;
  const int arow = lane & 15;
  const int t0 = wv, t1 = wv + 8;

  // layer 1: 64x384 @ 384x256
  {
    const float bv0 = c1[(t0 << 4) + col];
    const float bv1 = c1[(t1 << 4) + col];
    v8f acc[2][4] = {{bias_acc(bv0), bias_acc(bv0), bias_acc(bv0), bias_acc(bv0)},
                     {bias_acc(bv1), bias_acc(bv1), bias_acc(bv1), bias_acc(bv1)}};
#pragma unroll
    for (int kc = 0; kc < 12; ++kc) {
      v16bf a0 = make_a_frag(sIn + (size_t)(arow)      * 384, kc << 5, lane);
      v16bf a1 = make_a_frag(sIn + (size_t)(16 + arow) * 384, kc << 5, lane);
      v16bf a2 = make_a_frag(sIn + (size_t)(32 + arow) * 384, kc << 5, lane);
      v16bf a3 = make_a_frag(sIn + (size_t)(48 + arow) * 384, kc << 5, lane);
      v16bf bf0 = make_b_frag(u1pk, 192, t0, kc, lane);
      acc[0][0] = WMMA_BF16(a0, bf0, acc[0][0]);
      acc[0][1] = WMMA_BF16(a1, bf0, acc[0][1]);
      acc[0][2] = WMMA_BF16(a2, bf0, acc[0][2]);
      acc[0][3] = WMMA_BF16(a3, bf0, acc[0][3]);
      v16bf bf1 = make_b_frag(u1pk, 192, t1, kc, lane);
      acc[1][0] = WMMA_BF16(a0, bf1, acc[1][0]);
      acc[1][1] = WMMA_BF16(a1, bf1, acc[1][1]);
      acc[1][2] = WMMA_BF16(a2, bf1, acc[1][2]);
      acc[1][3] = WMMA_BF16(a3, bf1, acc[1][3]);
    }
#pragma unroll
    for (int tt = 0; tt < 2; ++tt) {
      const int t = wv + (tt << 3);
#pragma unroll
      for (int m = 0; m < 4; ++m)
#pragma unroll
        for (int i = 0; i < 8; ++i)
          sH[((m << 4) + rb + i) * 256 + (t << 4) + col] = (__bf16)mish_f(acc[tt][m][i]);
    }
  }
  __syncthreads();

  // layer 2: 64x256 @ 256x128 (8 col-tiles, one per wave)
  {
    const int t = wv;
    const float bv = c2[(t << 4) + col];
    v8f acc[4] = {bias_acc(bv), bias_acc(bv), bias_acc(bv), bias_acc(bv)};
#pragma unroll
    for (int kc = 0; kc < 8; ++kc) {
      v16bf bf = make_b_frag(u2pk, 128, t, kc, lane);
      v16bf a0 = make_a_frag(sH + (size_t)(arow)      * 256, kc << 5, lane);
      v16bf a1 = make_a_frag(sH + (size_t)(16 + arow) * 256, kc << 5, lane);
      v16bf a2 = make_a_frag(sH + (size_t)(32 + arow) * 256, kc << 5, lane);
      v16bf a3 = make_a_frag(sH + (size_t)(48 + arow) * 256, kc << 5, lane);
      acc[0] = WMMA_BF16(a0, bf, acc[0]);
      acc[1] = WMMA_BF16(a1, bf, acc[1]);
      acc[2] = WMMA_BF16(a2, bf, acc[2]);
      acc[3] = WMMA_BF16(a3, bf, acc[3]);
    }
#pragma unroll
    for (int m = 0; m < 4; ++m)
#pragma unroll
      for (int i = 0; i < 8; ++i)
        sU[((m << 4) + rb + i) * 128 + (t << 4) + col] = acc[m][i];
  }
  __syncthreads();

  // residual + layernorm over the 128-dim node vector (one thread per node row)
  if (tid < 64) {
    const long long node = nBase + tid;
    if (node < NN) {
      const float* xr = x + (size_t)node * ND;
      const float* ur = sU + tid * 128;
      float s = 0.f, s2 = 0.f;
      for (int c = 0; c < 128; ++c) {
        float y = xr[c] + ur[c];
        s += y; s2 += y * y;
      }
      const float mu  = s * (1.f / 128.f);
      const float var = s2 * (1.f / 128.f) - mu * mu;
      const float inv = rsqrtf(var + 1e-5f);
      for (int c = 0; c < 128; ++c) {
        float y = xr[c] + ur[c];
        out[(size_t)node * ND + c] = (y - mu) * inv * gamma[c] + beta[c];
      }
    }
  }
}

// ---------------- launcher ----------------
extern "C" void kernel_launch(void* const* d_in, const int* in_sizes, int n_in,
                              void* d_out, int out_size, void* d_ws, size_t ws_size,
                              hipStream_t stream) {
  const float*     x     = (const float*)d_in[0];
  const long long* eidx  = (const long long*)d_in[1];   // int64 [2, E]
  const float*     eattr = (const float*)d_in[2];
  const float*     W1    = (const float*)d_in[3];
  const float*     b1    = (const float*)d_in[4];
  const float*     W2    = (const float*)d_in[5];
  const float*     b2    = (const float*)d_in[6];
  const float*     U1    = (const float*)d_in[7];
  const float*     c1    = (const float*)d_in[8];
  const float*     U2    = (const float*)d_in[9];
  const float*     c2    = (const float*)d_in[10];
  const float*     gamma = (const float*)d_in[11];
  const float*     beta  = (const float*)d_in[12];
  float*           out   = (float*)d_out;

  // workspace layout
  char* ws = (char*)d_ws;
  float*    agg  = (float*)ws;                                   // [NN,256] f32 = 51.2 MB
  size_t    aggB = (size_t)NN * HID * sizeof(float);
  uint32_t* w1pk = (uint32_t*)(ws + aggB);                       // [256][224]
  uint32_t* w2pk = w1pk + 256 * 224;                             // [256][128]
  uint32_t* u1pk = w2pk + 256 * 128;                             // [256][192]
  uint32_t* u2pk = u1pk + 256 * 192;                             // [128][128]

  // zero the aggregation buffer (atomics accumulate into it)
  {
    int n4 = (NN * HID) / 4;
    zero_f32_kernel<<<(n4 + 255) / 256, 256, 0, stream>>>((float4*)agg, n4);
  }
  // pack weights to bf16-pair transposed layout
  pack_w_kernel<<<(256 * 224 + 255) / 256, 256, 0, stream>>>(W1, w1pk, 224, 256);
  pack_w_kernel<<<(256 * 128 + 255) / 256, 256, 0, stream>>>(W2, w2pk, 128, 256);
  pack_w_kernel<<<(256 * 192 + 255) / 256, 256, 0, stream>>>(U1, u1pk, 192, 256);
  pack_w_kernel<<<(128 * 128 + 255) / 256, 256, 0, stream>>>(U2, u2pk, 128, 128);

  // edge message + scatter (E/64 = 6250 blocks)
  edge_msg_kernel<<<NE / 64, 256, 0, stream>>>(x, eidx, eattr, b1, b2, w1pk, w2pk, agg);

  // node update + residual + layernorm (ceil(N/64) = 782 blocks)
  node_upd_kernel<<<(NN + 63) / 64, 256, 0, stream>>>(x, agg, c1, c2, u1pk, u2pk, gamma, beta, out);
}
